// GPT_88897233092662
// MI455X (gfx1250) — matrix operand order repaired and checked by reading, not compiled, not run on hardware
//
#include <hip/hip_runtime.h>
#include <math.h>

// ---------------------------------------------------------------------------
// GPT-2 small forward on gfx1250 (MI455X), wave32 + WMMA f32_16x16x32_f16,
// TDM (tensor_load_to_lds) staging of GEMM A tiles, global_prefetch on B.
// Residual stream fp32; GEMM inputs f16; fp32 accumulation.
// ---------------------------------------------------------------------------

typedef _Float16     v16h  __attribute__((ext_vector_type(16)));
typedef float        v8f   __attribute__((ext_vector_type(8)));
typedef unsigned int u32x4 __attribute__((ext_vector_type(4)));
typedef int          i32x8 __attribute__((ext_vector_type(8)));
typedef int          i32x4 __attribute__((ext_vector_type(4)));

#define WMMA_F16(A_, B_, C_) \
  __builtin_amdgcn_wmma_f32_16x16x32_f16(false, (A_), false, (B_), (short)0, (C_), false, false)

#if __has_builtin(__builtin_amdgcn_tensor_load_to_lds) && \
    __has_builtin(__builtin_amdgcn_s_wait_tensorcnt)
#define USE_TDM 1
#else
#define USE_TDM 0
#endif

// LDS byte offset of a __shared__ object (AS(3) pointers are 32-bit offsets).
#define LDS_OFFSET(p) \
  ((unsigned)(unsigned long long)(__attribute__((address_space(3))) const void*)(const void*)(p))

// A fragment: 16x32 f16 row-major, row stride lda (in halves).
// Lane l holds row (l&15); K elems kh..kh+15, kh = 16*(l>>4).
static __device__ __forceinline__ v16h load_frag_a(const _Float16* __restrict__ A, int lda) {
  const int l = threadIdx.x & 31;
  const _Float16* row = A + (l & 15) * lda;
  const int kh = (l >> 4) * 16;
  v16h a;
#pragma unroll
  for (int j = 0; j < 8; ++j) { a[j] = row[kh + j]; a[8 + j] = row[kh + 8 + j]; }
  return a;
}

// B fragment: 32x16, memory holds B^T rows: BT[n][k], row stride ldt (halves).
// Lane l holds col (l&15); K elems 16*(l>>4)+j  -> contiguous per lane.
static __device__ __forceinline__ v16h load_frag_bt(const _Float16* __restrict__ BT, int ldt) {
  const int l = threadIdx.x & 31;
  const _Float16* row = BT + (size_t)(l & 15) * ldt + (l >> 4) * 16;
  v16h b;
#pragma unroll
  for (int j = 0; j < 16; ++j) b[j] = row[j];
  return b;
}

// ---------------------------------------------------------------------------
// Embedding: h[b,t,:] = wte[x[b,t],:] + wpe[t,:]   (fp32 residual stream)
// ---------------------------------------------------------------------------
__global__ __launch_bounds__(256) void embed_kernel(
    const int* __restrict__ x, const float* __restrict__ wte,
    const float* __restrict__ wpe, float* __restrict__ h,
    int T, int C, int n) {
  int i = blockIdx.x * 256 + threadIdx.x;
  if (i >= n) return;
  int c = i % C;
  int bt = i / C;
  int t = bt % T;
  int tok = x[bt];
  h[i] = wte[(size_t)tok * C + c] + wpe[(size_t)t * C + c];
}

// ---------------------------------------------------------------------------
// LayerNorm: one row per block (256 threads, 8 waves); output f16.
// ---------------------------------------------------------------------------
__global__ __launch_bounds__(256) void layernorm_kernel(
    const float* __restrict__ hin, const float* __restrict__ w,
    const float* __restrict__ b, _Float16* __restrict__ out, int C) {
  const int row = blockIdx.x;
  const float* xr = hin + (size_t)row * C;
  __shared__ float red[8];
  const int lane = threadIdx.x & 31, wave = threadIdx.x >> 5;

  float s = 0.f;
  for (int c = threadIdx.x; c < C; c += 256) s += xr[c];
#pragma unroll
  for (int m = 16; m; m >>= 1) s += __shfl_xor(s, m, 32);
  if (lane == 0) red[wave] = s;
  __syncthreads();
  float mean = 0.f;
#pragma unroll
  for (int i = 0; i < 8; ++i) mean += red[i];
  mean /= (float)C;
  __syncthreads();

  float v = 0.f;
  for (int c = threadIdx.x; c < C; c += 256) { float d = xr[c] - mean; v += d * d; }
#pragma unroll
  for (int m = 16; m; m >>= 1) v += __shfl_xor(v, m, 32);
  if (lane == 0) red[wave] = v;
  __syncthreads();
  float var = 0.f;
#pragma unroll
  for (int i = 0; i < 8; ++i) var += red[i];
  var /= (float)C;
  const float rstd = rsqrtf(var + 1e-5f);

  _Float16* o = out + (size_t)row * C;
  for (int c = threadIdx.x; c < C; c += 256)
    o[c] = (_Float16)((xr[c] - mean) * rstd * w[c] + b[c]);
}

// ---------------------------------------------------------------------------
// GEMM: D[M,N] = A[M,K](f16) * Bw[K,N](f32->f16) + bias
//  MODE 0: out16 = D            (qkv)
//  MODE 1: out16 = gelu(D)      (mlp fc, exact gelu)
//  MODE 2: outf  += D           (residual adds: attn proj, mlp proj)
// Tiles: 128x64x32, 256 threads = 8 waves, wave w owns rows [16w,16w+16).
// A tile staged by TDM (tensor_load_to_lds) issued by wave 0; B tile staged
// by all threads with fp32->f16 convert + transpose; next B tile prefetched.
// ---------------------------------------------------------------------------
template <int MODE>
__global__ __launch_bounds__(256) void gemm_kernel(
    const _Float16* __restrict__ A, const float* __restrict__ Bw,
    const float* __restrict__ bias, _Float16* __restrict__ out16,
    float* __restrict__ outf, int M, int N, int K) {
  constexpr int MT = 128, NT = 64, KT = 32;
  __shared__ _Float16 As[MT][KT];
  __shared__ _Float16 Bs[NT][KT];  // holds B^T
  const int tid = threadIdx.x, wave = tid >> 5, lane = tid & 31;
  const int m0 = blockIdx.x * MT, n0 = blockIdx.y * NT;

#if USE_TDM
  const unsigned lds_a = LDS_OFFSET(&As[0][0]);
#endif

  v8f acc[4] = {};

  for (int k0 = 0; k0 < K; k0 += KT) {
#if USE_TDM
    if (wave == 0) {
      // D# group0: count=1 | lds_addr | global tile addr | type=2
      const unsigned long long ga =
          (unsigned long long)(const char*)(A + (size_t)m0 * K + k0);
      u32x4 g0;
      g0[0] = 1u;                                  // count=1, user mode
      g0[1] = lds_a;                               // lds_addr (bytes)
      g0[2] = (unsigned)ga;                        // global_addr[31:0]
      g0[3] = (unsigned)(ga >> 32) | (2u << 30);   // global_addr[56:32] | type=2
      // D# group1: data_size=2B; tensor K x M; tile 32 x 128; dim0 stride=K
      i32x8 g1;
      g1[0] = 0x10000;                             // data_size=1 (2 bytes)
      g1[1] = (int)((K & 0xFFFF) << 16);           // tensor_dim0[15:0]
      g1[2] = (int)(((unsigned)K >> 16) & 0xFFFF) |
              (int)((M & 0xFFFF) << 16);           // tensor_dim0[31:16] | dim1[15:0]
      g1[3] = (int)(((unsigned)M >> 16) & 0xFFFF) |
              (KT << 16);                          // tensor_dim1[31:16] | tile_dim0=32
      g1[4] = MT;                                  // tile_dim1=128, tile_dim2=0
      g1[5] = K;                                   // tensor_dim0_stride[31:0]
      g1[6] = 0;
      g1[7] = 0;
      i32x4 g2 = {0, 0, 0, 0};
      i32x4 g3 = {0, 0, 0, 0};
      i32x8 g4 = {0, 0, 0, 0, 0, 0, 0, 0};
      __builtin_amdgcn_tensor_load_to_lds(g0, g1, g2, g3, g4, 0);
    }
#else
    {  // manual stage A tile: 128 rows x 32 halves (lda == K)
      int row = tid >> 1, seg = (tid & 1) * 16;
      const _Float16* src = A + (size_t)(m0 + row) * K + k0 + seg;
      *(uint4*)&As[row][seg]     = *(const uint4*)src;
      *(uint4*)&As[row][seg + 8] = *(const uint4*)(src + 8);
    }
#endif
    {  // stage B tile: 32(k) x 64(n) fp32 -> f16, transposed into Bs[n][k]
      int k = tid >> 3, nc = (tid & 7) * 8;
      const float* src = Bw + (size_t)(k0 + k) * N + n0 + nc;
      const float4* s4 = (const float4*)src;
      float4 f0 = s4[0], f1 = s4[1];
      Bs[nc + 0][k] = (_Float16)f0.x; Bs[nc + 1][k] = (_Float16)f0.y;
      Bs[nc + 2][k] = (_Float16)f0.z; Bs[nc + 3][k] = (_Float16)f0.w;
      Bs[nc + 4][k] = (_Float16)f1.x; Bs[nc + 5][k] = (_Float16)f1.y;
      Bs[nc + 6][k] = (_Float16)f1.z; Bs[nc + 7][k] = (_Float16)f1.w;
      if (k0 + KT < K)  // global_prefetch next weight tile
        __builtin_prefetch(src + (size_t)KT * N, 0, 3);
    }
#if USE_TDM
    if (wave == 0) __builtin_amdgcn_s_wait_tensorcnt(0);
#endif
    __syncthreads();
    v16h a = load_frag_a(&As[wave * 16][0], KT);
#pragma unroll
    for (int f = 0; f < 4; ++f) {
      v16h b = load_frag_bt(&Bs[f * 16][0], KT);
      acc[f] = WMMA_F16(a, b, acc[f]);
    }
    __syncthreads();
  }

  const int mb = (lane >> 4) * 8, nn = lane & 15;
#pragma unroll
  for (int f = 0; f < 4; ++f) {
    const int gn = n0 + f * 16 + nn;
    const float bv = bias[gn];
#pragma unroll
    for (int r = 0; r < 8; ++r) {
      const int gm = m0 + wave * 16 + mb + r;
      float v = acc[f][r] + bv;
      if (MODE == 1) v = 0.5f * v * (1.0f + erff(v * 0.70710678118f));
      if (MODE == 2) outf[(size_t)gm * N + gn] += v;
      else           out16[(size_t)gm * N + gn] = (_Float16)v;
    }
  }
}

// ---------------------------------------------------------------------------
// Flash attention: one wave per (b, head, 16-row q tile). Online softmax,
// S = Q K^T and P V on WMMA; P re-laid C->A layout via per-wave LDS bounce.
// qkv layout: [b][t][3C], q at +0, k at +C, v at +2C; head dim D = 64.
// ---------------------------------------------------------------------------
__global__ __launch_bounds__(256) void attn_kernel(
    const _Float16* __restrict__ qkv, _Float16* __restrict__ out,
    int B, int T, int C, int Hn, float scale) {
  const int D = C / Hn;  // 64
  __shared__ _Float16 plds[8][16 * 32];
  const int wave = threadIdx.x >> 5, lane = threadIdx.x & 31;
  const int tile = blockIdx.x * 8 + wave;
  const int ntq = T / 16;
  const int qt = tile % ntq;
  const int bh = tile / ntq;
  const int hh = bh % Hn;
  const int b  = bh / Hn;

  const size_t rs = (size_t)3 * C;  // row stride in halves
  const _Float16* qbase = qkv + (size_t)b * T * rs + (size_t)hh * D;
  const _Float16* kbase = qbase + C;
  const _Float16* vbase = qbase + 2 * C;

  const int m    = lane & 15;
  const int kh8  = (lane >> 4) * 8;
  const int kh16 = (lane >> 4) * 16;
  const int mb   = (lane >> 4) * 8;
  const int nn   = lane & 15;

  // Q fragments (16 rows x 64 dims = two 16x32 A-frags)
  v16h aq[2];
#pragma unroll
  for (int dc = 0; dc < 2; ++dc) {
    const _Float16* qrow = qbase + (size_t)(qt * 16 + m) * rs + dc * 32;
#pragma unroll
    for (int j = 0; j < 8; ++j) { aq[dc][j] = qrow[kh16 + j]; aq[dc][8 + j] = qrow[kh16 + 8 + j]; }
  }

  v8f acc[4] = {};
  float mi[8], li[8];
#pragma unroll
  for (int r = 0; r < 8; ++r) { mi[r] = -1e30f; li[r] = 0.f; }

  const int kmax = qt * 16 + 15;  // last valid key (causal)
  for (int ks = 0; ks <= kmax; ks += 32) {
    // ---- S = Q K^T over 32 keys: two n-frags, accumulate over 2 d-chunks
    v8f sf[2] = {};
#pragma unroll
    for (int nf = 0; nf < 2; ++nf) {
#pragma unroll
      for (int dc = 0; dc < 2; ++dc) {
        // B[d][n] = K[ks+nf*16+n][dc*32 + 16*(lane>>4) + j]  (contiguous per lane)
        const _Float16* kr = kbase + (size_t)(ks + nf * 16 + nn) * rs + dc * 32 + kh16;
        v16h bk;
#pragma unroll
        for (int j = 0; j < 16; ++j) bk[j] = kr[j];
        sf[nf] = WMMA_F16(aq[dc], bk, sf[nf]);
      }
    }

    // ---- scale, causal mask, online softmax update
#pragma unroll
    for (int r = 0; r < 8; ++r) {
      const int tq = qt * 16 + mb + r;
      float s0 = sf[0][r] * scale; if (ks + nn > tq)      s0 = -1e30f;
      float s1 = sf[1][r] * scale; if (ks + 16 + nn > tq) s1 = -1e30f;
      float mr = fmaxf(s0, s1);
#pragma unroll
      for (int sh = 1; sh < 16; sh <<= 1) mr = fmaxf(mr, __shfl_xor(mr, sh, 32));
      const float mnew = fmaxf(mi[r], mr);
      const float alpha = __expf(mi[r] - mnew);
      const float p0 = __expf(s0 - mnew);
      const float p1 = __expf(s1 - mnew);
      sf[0][r] = p0; sf[1][r] = p1;
      float rsum = p0 + p1;
#pragma unroll
      for (int sh = 1; sh < 16; sh <<= 1) rsum += __shfl_xor(rsum, sh, 32);
      li[r] = li[r] * alpha + rsum;
      mi[r] = mnew;
#pragma unroll
      for (int f = 0; f < 4; ++f) acc[f][r] *= alpha;
    }

    // ---- P (C-layout) -> LDS -> A-layout
    _Float16* pl = plds[wave];
#pragma unroll
    for (int r = 0; r < 8; ++r) {
      pl[(mb + r) * 32 + nn]      = (_Float16)sf[0][r];
      pl[(mb + r) * 32 + 16 + nn] = (_Float16)sf[1][r];
    }
    asm volatile("s_wait_dscnt 0" ::: "memory");
    v16h ap;
    {
      const _Float16* pr = pl + m * 32;
#pragma unroll
      for (int j = 0; j < 8; ++j) { ap[j] = pr[kh16 + j]; ap[8 + j] = pr[kh16 + 8 + j]; }
    }

    // ---- acc += P V   (V is 32x64: four B-frags, k = keys)
#pragma unroll
    for (int f = 0; f < 4; ++f) {
      v16h bv;
#pragma unroll
      for (int j = 0; j < 16; ++j)
        bv[j] = vbase[(size_t)(ks + kh16 + j) * rs + f * 16 + nn];
      acc[f] = WMMA_F16(ap, bv, acc[f]);
    }
  }

  // ---- epilogue: O = acc / l_i  -> out[b][t][hh*D + dv] as f16
#pragma unroll
  for (int f = 0; f < 4; ++f) {
#pragma unroll
    for (int r = 0; r < 8; ++r) {
      const int tq = qt * 16 + mb + r;
      const float o = acc[f][r] / li[r];
      out[(size_t)(b * T + tq) * C + hh * D + f * 16 + nn] = (_Float16)o;
    }
  }
}

// ---------------------------------------------------------------------------
// Logits GEMV: one wave per (b, v); lanes stride C, coalesced wte reads.
// ---------------------------------------------------------------------------
__global__ __launch_bounds__(256) void logits_kernel(
    const _Float16* __restrict__ xn, const float* __restrict__ wte,
    float* __restrict__ out, int T, int C, int V) {
  const int wave = threadIdx.x >> 5, lane = threadIdx.x & 31;
  const int idx = blockIdx.x * 8 + wave;
  const int b = idx / V, v = idx % V;
  const _Float16* xr = xn + (size_t)(b * T + (T - 1)) * C;
  const float* wr = wte + (size_t)v * C;
  float s = 0.f;
  for (int c = lane; c < C; c += 32) s += (float)xr[c] * wr[c];
#pragma unroll
  for (int sh = 16; sh; sh >>= 1) s += __shfl_xor(s, sh, 32);
  if (lane == 0) out[(size_t)b * V + v] = s;
}

// ---------------------------------------------------------------------------
extern "C" void kernel_launch(void* const* d_in, const int* in_sizes, int n_in,
                              void* d_out, int out_size, void* d_ws, size_t ws_size,
                              hipStream_t stream) {
  const int*   x     = (const int*)  d_in[0];
  const float* wte   = (const float*)d_in[1];
  const float* wpe   = (const float*)d_in[2];
  const float* ln1_w = (const float*)d_in[3];
  const float* ln1_b = (const float*)d_in[4];
  const float* wqkv  = (const float*)d_in[5];
  const float* bqkv  = (const float*)d_in[6];
  const float* wo    = (const float*)d_in[7];
  const float* bo    = (const float*)d_in[8];
  const float* ln2_w = (const float*)d_in[9];
  const float* ln2_b = (const float*)d_in[10];
  const float* wfc   = (const float*)d_in[11];
  const float* bfc   = (const float*)d_in[12];
  const float* wproj = (const float*)d_in[13];
  const float* bproj = (const float*)d_in[14];
  const float* lnf_w = (const float*)d_in[15];
  const float* lnf_b = (const float*)d_in[16];

  const int B = 8, T = 1024, C = 768, Hn = 12, L = 12, V = 50257;
  const int BT = B * T;  // 8192

  // workspace layout (all offsets well aligned)
  char* ws = (char*)d_ws;
  float*    h     = (float*)ws;                                        // BT*C fp32
  _Float16* xn16  = (_Float16*)(ws + (size_t)BT * C * 4);              // BT*C f16
  _Float16* qkv16 = (_Float16*)((char*)xn16  + (size_t)BT * C * 2);    // BT*3C f16
  _Float16* att16 = (_Float16*)((char*)qkv16 + (size_t)BT * 3 * C * 2);// BT*C f16
  _Float16* fc16  = (_Float16*)((char*)att16 + (size_t)BT * C * 2);    // BT*4C f16

  const int n_embed = BT * C;
  embed_kernel<<<(n_embed + 255) / 256, 256, 0, stream>>>(x, wte, wpe, h, T, C, n_embed);

  const dim3 gq(BT / 128, 3 * C / 64);   // qkv
  const dim3 gc(BT / 128, C / 64);       // proj / mlp-proj output (N=C)
  const dim3 gf(BT / 128, 4 * C / 64);   // fc
  const int attn_blocks = (B * Hn * (T / 16)) / 8;

  for (int l = 0; l < L; ++l) {
    layernorm_kernel<<<BT, 256, 0, stream>>>(h, ln1_w + l * C, ln1_b + l * C, xn16, C);
    gemm_kernel<0><<<gq, 256, 0, stream>>>(xn16, wqkv + (size_t)l * C * 3 * C,
                                           bqkv + (size_t)l * 3 * C, qkv16, nullptr,
                                           BT, 3 * C, C);
    attn_kernel<<<attn_blocks, 256, 0, stream>>>(qkv16, att16, B, T, C, Hn, 0.125f);
    gemm_kernel<2><<<gc, 256, 0, stream>>>(att16, wo + (size_t)l * C * C,
                                           bo + (size_t)l * C, nullptr, h,
                                           BT, C, C);
    layernorm_kernel<<<BT, 256, 0, stream>>>(h, ln2_w + l * C, ln2_b + l * C, xn16, C);
    gemm_kernel<1><<<gf, 256, 0, stream>>>(xn16, wfc + (size_t)l * C * 4 * C,
                                           bfc + (size_t)l * 4 * C, fc16, nullptr,
                                           BT, 4 * C, C);
    gemm_kernel<2><<<gc, 256, 0, stream>>>(fc16, wproj + (size_t)l * 4 * C * C,
                                           bproj + (size_t)l * C, nullptr, h,
                                           BT, C, 4 * C);
  }

  layernorm_kernel<<<BT, 256, 0, stream>>>(h, lnf_w, lnf_b, xn16, C);
  logits_kernel<<<V, 256, 0, stream>>>(xn16, wte, (float*)d_out, T, C, V);
}
